// Block_6511170421026
// MI455X (gfx1250) — matrix operand order repaired and checked by reading, not compiled, not run on hardware
//
#include <hip/hip_runtime.h>

// ---------------------------------------------------------------------------
// Types: use raw ext_vector types (trivial, union-safe) instead of HIP structs
// ---------------------------------------------------------------------------
typedef __attribute__((ext_vector_type(16))) __bf16        bf16x16;
typedef __attribute__((ext_vector_type(8)))  float         f32x8;
typedef __attribute__((ext_vector_type(4)))  unsigned int  u32x4;
typedef __attribute__((ext_vector_type(4)))  float         f32x4;

union FragB16 {
    bf16x16 v;
    u32x4   q[2];
};

__device__ __forceinline__ unsigned short f32_to_bf16(float f) {
    unsigned int u = __float_as_uint(f);
    u += 0x7fffu + ((u >> 16) & 1u);   // round-to-nearest-even
    return (unsigned short)(u >> 16);
}

// A matrix (16x32 bf16) per-lane layout (ISA 7.12.2):
// lane = 16*half + m ; element e -> k = 16*(e>>3) + 8*half + (e&7)
// => two contiguous 16B chunks at k0+8*half and k0+16+8*half.
__device__ __forceinline__ void load_a_frag(FragB16& f, const unsigned short* rowp, int half) {
    f.q[0] = *(const u32x4*)(rowp + 8 * half);
    f.q[1] = *(const u32x4*)(rowp + 16 + 8 * half);
}

// B matrix (32x16 bf16) per-lane layout (same k order as A):
// lane = 16*half + n ; element e -> k = 16*(e>>3) + 8*half + (e&7)
// => two contiguous 16B chunks at k0+8*half and k0+16+8*half.
__device__ __forceinline__ void load_b_frag(FragB16& f, const unsigned short* rowp, int half) {
    f.q[0] = *(const u32x4*)(rowp + 8 * half);
    f.q[1] = *(const u32x4*)(rowp + 16 + 8 * half);
}

__device__ __forceinline__ f32x8 wmma_bf16(const FragB16& a, const FragB16& b, f32x8 c) {
    return __builtin_amdgcn_wmma_f32_16x16x32_bf16(false, a.v, false, b.v,
                                                   (short)0, c, false, false);
}

// ---------------------------------------------------------------------------
// Dimensions (compile-time constants for this block)
// ---------------------------------------------------------------------------
#define DIMD    768
#define D3      2304
#define HIDDEN  3072
#define NHEADS  12
#define HEADD   64
#define SEQ     2048
#define BATCH   4
#define TOKENS  (BATCH * SEQ)   // 8192

// ---------------------------------------------------------------------------
// Kernel 1: f32 -> bf16 elementwise (weights)
// ---------------------------------------------------------------------------
__global__ void cvt_bf16_kernel(const float* __restrict__ src,
                                unsigned short* __restrict__ dst, int n) {
    int i = blockIdx.x * blockDim.x + threadIdx.x;
    int stride = gridDim.x * blockDim.x;
    for (; i < n; i += stride) dst[i] = f32_to_bf16(src[i]);
}

// ---------------------------------------------------------------------------
// Kernel 2: LayerNorm (f32 in) -> bf16 out.  One wave32 per token row.
// ---------------------------------------------------------------------------
__global__ __launch_bounds__(256)
void ln_bf16_kernel(const float* __restrict__ x, const float* __restrict__ g,
                    const float* __restrict__ bta, unsigned short* __restrict__ out,
                    int rows) {
    int row  = blockIdx.x * (blockDim.x >> 5) + (threadIdx.x >> 5);
    int lane = threadIdx.x & 31;
    if (row >= rows) return;

    const f32x4* x4 = (const f32x4*)(x + (size_t)row * DIMD);
    f32x4 v[6];
    float s = 0.f, ss = 0.f;
#pragma unroll
    for (int j = 0; j < 6; ++j) {
        v[j] = x4[lane + 32 * j];
        s  += v[j].x + v[j].y + v[j].z + v[j].w;
        ss += v[j].x * v[j].x + v[j].y * v[j].y + v[j].z * v[j].z + v[j].w * v[j].w;
    }
#pragma unroll
    for (int m = 16; m >= 1; m >>= 1) {
        s  += __shfl_xor(s,  m, 32);
        ss += __shfl_xor(ss, m, 32);
    }
    const float inv_n = 1.0f / (float)DIMD;
    float mu   = s * inv_n;
    float var  = ss * inv_n - mu * mu;
    float rstd = rsqrtf(var + 1e-6f);

    unsigned short* orow = out + (size_t)row * DIMD;
    const f32x4* g4 = (const f32x4*)g;
    const f32x4* b4 = (const f32x4*)bta;
#pragma unroll
    for (int j = 0; j < 6; ++j) {
        int   i4 = lane + 32 * j;
        f32x4 gv = g4[i4];
        f32x4 bv = b4[i4];
        int   i0 = i4 * 4;
        orow[i0 + 0] = f32_to_bf16((v[j].x - mu) * rstd * gv.x + bv.x);
        orow[i0 + 1] = f32_to_bf16((v[j].y - mu) * rstd * gv.y + bv.y);
        orow[i0 + 2] = f32_to_bf16((v[j].z - mu) * rstd * gv.z + bv.z);
        orow[i0 + 3] = f32_to_bf16((v[j].w - mu) * rstd * gv.w + bv.w);
    }
}

// ---------------------------------------------------------------------------
// Kernel 3: bf16 WMMA GEMM   C[M,N] = A[M,K] * W[N,K]^T + bias (+gelu)(+res)
// Block: 256 threads = 8 waves, block tile 128(M) x 128(N).
// Wave tile 32(M) x 64(N): 2 A-frags x 4 B-frags -> 8 WMMAs per K=32 step
// (1.5 b128 loads per WMMA; 8 independent WMMAs to hide L2 latency).
// ---------------------------------------------------------------------------
__device__ __forceinline__ float gelu_exact(float v) {
    return 0.5f * v * (1.0f + erff(v * 0.70710678118654752f));
}

template <int ACT, bool RES, bool WRITEF>
__global__ __launch_bounds__(256)
void gemm_bf16_kernel(const unsigned short* __restrict__ A,
                      const unsigned short* __restrict__ W,
                      const float* __restrict__ bias,
                      const float* __restrict__ residual,
                      float* __restrict__ outF,
                      unsigned short* __restrict__ outB,
                      int M, int N, int K) {
    const int lane   = threadIdx.x & 31;
    const int lane16 = lane & 15;
    const int half   = lane >> 4;
    const int w      = threadIdx.x >> 5;   // 0..7
    const int wm     = w & 3;              // 4 M-slots of 32 rows
    const int wn     = w >> 2;             // 2 N-slots of 64 cols

    const int m0 = blockIdx.y * 128 + wm * 32;
    const int n0 = blockIdx.x * 128 + wn * 64;

    f32x8 acc[2][4] = {};

    for (int k0 = 0; k0 < K; k0 += 32) {
        FragB16 af[2], bf[4];
#pragma unroll
        for (int mt = 0; mt < 2; ++mt)
            load_a_frag(af[mt], A + (size_t)(m0 + mt * 16 + lane16) * K + k0, half);
#pragma unroll
        for (int nt = 0; nt < 4; ++nt)
            load_b_frag(bf[nt], W + (size_t)(n0 + nt * 16 + lane16) * K + k0, half);
#pragma unroll
        for (int mt = 0; mt < 2; ++mt)
#pragma unroll
            for (int nt = 0; nt < 4; ++nt)
                acc[mt][nt] = wmma_bf16(af[mt], bf[nt], acc[mt][nt]);
    }

    // Epilogue. C layout: VGPR r -> row m = r + 8*half, col n = lane16.
#pragma unroll
    for (int mt = 0; mt < 2; ++mt) {
#pragma unroll
        for (int nt = 0; nt < 4; ++nt) {
            int   n  = n0 + nt * 16 + lane16;
            float bv = bias[n];
#pragma unroll
            for (int r = 0; r < 8; ++r) {
                int   m = m0 + mt * 16 + 8 * half + r;
                float v = acc[mt][nt][r] + bv;
                if (ACT == 1) v = gelu_exact(v);
                if (RES) v += residual[(size_t)m * N + n];
                if (WRITEF) outF[(size_t)m * N + n] = v;
                else        outB[(size_t)m * N + n] = f32_to_bf16(v);
            }
        }
    }
}

// ---------------------------------------------------------------------------
// Kernel 4: flash attention.  grid = (B*H, SEQ/64), block = 128 (4 waves).
// Each wave: one 16-row query tile; online softmax over key chunks of 32.
// qkv is bf16 [TOKENS, 3*DIMD] with [q | k | v] each [H, 64] per token.
// ---------------------------------------------------------------------------
__global__ __launch_bounds__(128)
void attention_kernel(const unsigned short* __restrict__ qkv,
                      unsigned short* __restrict__ attn_out) {
    const int bh = blockIdx.x;
    const int b  = bh / NHEADS;
    const int h  = bh % NHEADS;
    const int wv     = threadIdx.x >> 5;
    const int lane   = threadIdx.x & 31;
    const int lane16 = lane & 15;
    const int half   = lane >> 4;
    const int q0     = blockIdx.y * 64 + wv * 16;

    const size_t tokBase = (size_t)b * SEQ;
    const unsigned short* Q  = qkv + tokBase * D3 + 0 * DIMD + h * HEADD;
    const unsigned short* Kp = qkv + tokBase * D3 + 1 * DIMD + h * HEADD;
    const unsigned short* Vp = qkv + tokBase * D3 + 2 * DIMD + h * HEADD;

    // Q fragments (16 x 64 = two K=32 chunks), loaded once.
    FragB16 qf[2];
#pragma unroll
    for (int kk = 0; kk < 2; ++kk)
        load_a_frag(qf[kk], Q + (size_t)(q0 + lane16) * D3 + kk * 32, half);

    f32x8 o[4] = {};
    float mrow[8], lrow[8];
#pragma unroll
    for (int r = 0; r < 8; ++r) { mrow[r] = -1e30f; lrow[r] = 0.f; }

    __shared__ unsigned short VtS[64 * 40];     // V chunk transposed: [d][key], stride 40
    __shared__ unsigned short Pb[4][16 * 40];   // per-wave P tile:    [m][key], stride 40
    unsigned short* Pw = Pb[wv];

    const float scale = 0.125f;                 // 64^-0.5

    for (int kc = 0; kc < SEQ / 32; ++kc) {
        const int key0 = kc * 32;

        __syncthreads();
        // Cooperative transpose of V chunk [32 keys x 64 d] -> VtS[d][key].
#pragma unroll
        for (int it = 0; it < 2; ++it) {
            int u    = threadIdx.x + it * 128;  // 0..255 uint4 slots
            int key  = u >> 3;
            int dblk = (u & 7) * 8;
            union { u32x4 q; unsigned short s[8]; } uu;
            uu.q = *(const u32x4*)(Vp + (size_t)(key0 + key) * D3 + dblk);
#pragma unroll
            for (int j = 0; j < 8; ++j) VtS[(dblk + j) * 40 + key] = uu.s[j];
        }
        __syncthreads();

        // Scores S[16 x 32] = Q (16x64) @ K^T; keys are the N dimension.
        f32x8 s0 = {}, s1 = {};
#pragma unroll
        for (int kk = 0; kk < 2; ++kk) {
            FragB16 kb0, kb1;
            load_b_frag(kb0, Kp + (size_t)(key0 + lane16)      * D3 + kk * 32, half);
            load_b_frag(kb1, Kp + (size_t)(key0 + 16 + lane16) * D3 + kk * 32, half);
            s0 = wmma_bf16(qf[kk], kb0, s0);
            s1 = wmma_bf16(qf[kk], kb1, s1);
        }

        // Online softmax update, per row m = r + 8*half (cols live across 16 lanes).
        float alpha_r[8];
#pragma unroll
        for (int r = 0; r < 8; ++r) {
            float a = s0[r] * scale;
            float c = s1[r] * scale;
            float t = fmaxf(a, c);
#pragma unroll
            for (int m = 8; m >= 1; m >>= 1) t = fmaxf(t, __shfl_xor(t, m, 32));
            float mn = fmaxf(mrow[r], t);
            float p0 = __expf(a - mn);
            float p1 = __expf(c - mn);
            float al = __expf(mrow[r] - mn);
            mrow[r] = mn;
            float rs = p0 + p1;
#pragma unroll
            for (int m = 8; m >= 1; m >>= 1) rs += __shfl_xor(rs, m, 32);
            lrow[r] = lrow[r] * al + rs;
            alpha_r[r] = al;
            Pw[(r + 8 * half) * 40 + lane16]      = f32_to_bf16(p0);
            Pw[(r + 8 * half) * 40 + 16 + lane16] = f32_to_bf16(p1);
        }
#pragma unroll
        for (int db = 0; db < 4; ++db)
#pragma unroll
            for (int r = 0; r < 8; ++r) o[db][r] *= alpha_r[r];

        __syncthreads();   // make P (cross-lane) and Vt visible

        // O += P (16x32) @ V (32x64): 4 WMMAs over d-blocks of 16.
        FragB16 pf;
        load_a_frag(pf, Pw + lane16 * 40, half);
#pragma unroll
        for (int db = 0; db < 4; ++db) {
            FragB16 vb;
            load_b_frag(vb, VtS + (db * 16 + lane16) * 40, half);
            o[db] = wmma_bf16(pf, vb, o[db]);
        }
    }

    // Normalize and write bf16 output [TOKENS, DIMD] at column h*64 + d.
#pragma unroll
    for (int r = 0; r < 8; ++r) {
        float inv = 1.0f / lrow[r];
        int   m   = q0 + 8 * half + r;
        size_t rowoff = (tokBase + m) * DIMD + h * HEADD;
#pragma unroll
        for (int db = 0; db < 4; ++db)
            attn_out[rowoff + db * 16 + lane16] = f32_to_bf16(o[db][r] * inv);
    }
}

// ---------------------------------------------------------------------------
// Host-side orchestration
// ---------------------------------------------------------------------------
extern "C" void kernel_launch(void* const* d_in, const int* in_sizes, int n_in,
                              void* d_out, int out_size, void* d_ws, size_t ws_size,
                              hipStream_t stream) {
    const float* x      = (const float*)d_in[0];
    const float* ln1_g  = (const float*)d_in[1];
    const float* ln1_b  = (const float*)d_in[2];
    const float* qkv_w  = (const float*)d_in[3];
    const float* qkv_b  = (const float*)d_in[4];
    const float* proj_w = (const float*)d_in[5];
    const float* proj_b = (const float*)d_in[6];
    const float* ln2_g  = (const float*)d_in[7];
    const float* ln2_b  = (const float*)d_in[8];
    const float* fc1_w  = (const float*)d_in[9];
    const float* fc1_b  = (const float*)d_in[10];
    const float* fc2_w  = (const float*)d_in[11];
    const float* fc2_b  = (const float*)d_in[12];
    float* out = (float*)d_out;

    // Workspace layout (shorts unless noted); hmid aliases qkv+attn region.
    unsigned short* ws    = (unsigned short*)d_ws;
    unsigned short* wqkv  = ws;                                   // 2304*768
    unsigned short* wproj = wqkv  + (size_t)D3 * DIMD;            // 768*768
    unsigned short* wfc1  = wproj + (size_t)DIMD * DIMD;          // 3072*768
    unsigned short* wfc2  = wfc1  + (size_t)HIDDEN * DIMD;        // 768*3072
    unsigned short* hln   = wfc2  + (size_t)DIMD * HIDDEN;        // 8192*768
    unsigned short* qkvb  = hln   + (size_t)TOKENS * DIMD;        // 8192*2304
    unsigned short* attn  = qkvb  + (size_t)TOKENS * D3;          // 8192*768
    unsigned short* hmid  = qkvb;                                 // 8192*3072 (alias)
    float*          x1    = (float*)(attn + (size_t)TOKENS * DIMD); // 8192*768 f32

    // 1. Convert weights to bf16 (deterministic every call).
    cvt_bf16_kernel<<<1024, 256, 0, stream>>>(qkv_w,  wqkv,  D3 * DIMD);
    cvt_bf16_kernel<<<1024, 256, 0, stream>>>(proj_w, wproj, DIMD * DIMD);
    cvt_bf16_kernel<<<1024, 256, 0, stream>>>(fc1_w,  wfc1,  HIDDEN * DIMD);
    cvt_bf16_kernel<<<1024, 256, 0, stream>>>(fc2_w,  wfc2,  DIMD * HIDDEN);

    // 2. LN1 -> bf16
    ln_bf16_kernel<<<TOKENS / 8, 256, 0, stream>>>(x, ln1_g, ln1_b, hln, TOKENS);

    // 3. qkv = hln @ qkv_w^T + qkv_b  (bf16 out)
    gemm_bf16_kernel<0, false, false><<<dim3(D3 / 128, TOKENS / 128), 256, 0, stream>>>(
        hln, wqkv, qkv_b, nullptr, nullptr, qkvb, TOKENS, D3, DIMD);

    // 4. attention (bf16 out)
    attention_kernel<<<dim3(BATCH * NHEADS, SEQ / 64), 128, 0, stream>>>(qkvb, attn);

    // 5. x1 = x + attn @ proj_w^T + proj_b  (f32 out)
    gemm_bf16_kernel<0, true, true><<<dim3(DIMD / 128, TOKENS / 128), 256, 0, stream>>>(
        attn, wproj, proj_b, x, x1, nullptr, TOKENS, DIMD, DIMD);

    // 6. LN2 -> bf16
    ln_bf16_kernel<<<TOKENS / 8, 256, 0, stream>>>(x1, ln2_g, ln2_b, hln, TOKENS);

    // 7. hmid = gelu(hln @ fc1_w^T + fc1_b)  (bf16 out)
    gemm_bf16_kernel<1, false, false><<<dim3(HIDDEN / 128, TOKENS / 128), 256, 0, stream>>>(
        hln, wfc1, fc1_b, nullptr, nullptr, hmid, TOKENS, HIDDEN, DIMD);

    // 8. out = x1 + hmid @ fc2_w^T + fc2_b  (f32 out)
    gemm_bf16_kernel<0, true, true><<<dim3(DIMD / 128, TOKENS / 128), 256, 0, stream>>>(
        hmid, wfc2, fc2_b, x1, out, nullptr, TOKENS, DIMD, HIDDEN);
}
